// MB_DIDL_6768868459152
// MI455X (gfx1250) — matrix operand order repaired and checked by reading, not compiled, not run on hardware
//
#include <hip/hip_runtime.h>
#include <hip/hip_bf16.h>

typedef __attribute__((ext_vector_type(16))) _Float16 v16h;
typedef __attribute__((ext_vector_type(8)))  float    v8f;

#define NTOT  100000
#define NUSER 40000
#define DD    64
#define RRR   3
#define KLAY  2
#define INTN  4
#define EDG   1000000
#define NTILE 6250   /* NTOT/16 */

__device__ __forceinline__ v8f wmma_f16(v16h a, v16h b, v8f c) {
    return __builtin_amdgcn_wmma_f32_16x16x32_f16(false, a, false, b, (short)0, c, false, false);
}

__device__ __forceinline__ float lrelu(float x) { return x > 0.f ? x : 0.01f * x; }

// ---------------------------------------------------------------------------
// Pack one 64x64 f32 weight matrix into f16 WMMA-B per-lane layout:
// pk[((et*2+kb)*32 + lane)*16 + h] = W[k][e]  (k = kb*32 + (h<8?h:h+8) + hi*8,
// e = et*16 + lo).  Each lane's v16h operand becomes one contiguous 32B load.
// ---------------------------------------------------------------------------
__global__ void pack_weights_kernel(const float* __restrict__ intent_W,
                                    const float* __restrict__ W_gcn,
                                    const float* __restrict__ W_att1,
                                    const float* __restrict__ W_att2,
                                    _Float16* __restrict__ pk) {
    int m = blockIdx.x;            // 0..7
    int t = threadIdx.x;           // 0..255 = (op, lane)
    const float* W;
    int trans = 0;
    if (m < 4)       W = intent_W + m * 4096;
    else if (m < 6)  W = W_gcn + (m - 4) * 4096;
    else if (m == 6) W = W_att1;
    else           { W = W_att2; trans = 1; }
    int op = t >> 5, lane = t & 31;
    int et = op >> 1, kb = op & 1;
    int lo = lane & 15, hi = lane >> 4;
    int e = et * 16 + lo;
    _Float16* dst = pk + ((size_t)m * 256 + op * 32 + lane) * 16;
    for (int h = 0; h < 16; ++h) {
        int k = kb * 32 + (h < 8 ? h : h + 8) + hi * 8;
        dst[h] = (_Float16)(trans ? W[e * 64 + k] : W[k * 64 + e]);
    }
}

// ---------------------------------------------------------------------------
// Fused: ego0 assembly -> global_emb/all_global init; gates (softmax) first,
// then per column-tile: 4 intent WMMA tiles -> immediate gate contraction ->
// ego/all_emb.  Only 4 accumulator tiles live at once (no spills).
// One wave = 16 nodes.
// ---------------------------------------------------------------------------
__global__ void fused_init_kernel(const float* __restrict__ uemb,
                                  const float* __restrict__ iemb,
                                  const float* __restrict__ beh,
                                  const v16h* __restrict__ pk,      // matrices 0..3
                                  const float* __restrict__ intent_b,
                                  const float* __restrict__ gate_W,
                                  const float* __restrict__ gate_b,
                                  float* __restrict__ global_emb,
                                  float* __restrict__ all_global,
                                  float* __restrict__ ego,
                                  float* __restrict__ all_emb) {
    __shared__ float x_lds[16 * 64];
    __shared__ float g_lds[3 * 16 * 4];
    int tile = blockIdx.x;
    if (tile >= NTILE) return;
    int lane = threadIdx.x;
    int base = tile * 16;
    const float* src; int srow;
    if (base < NUSER) { src = uemb; srow = base; } else { src = iemb; srow = base - NUSER; }
    for (int j = 0; j < 32; ++j) {
        int idx = j * 32 + lane;               // 0..1023 = m*64+d
        float v = src[srow * 64 + idx];
        x_lds[idx] = v;
        global_emb[base * 64 + idx] = v;
        all_global[base * 64 + idx] = v;
    }
    __syncthreads();
    // gates: softmax over intents, per (r, node) -- before any WMMA so the
    // accumulator live-range stays short
    for (int p = lane; p < 48; p += 32) {
        int n = p & 15, r = p >> 4;
        float lg[INTN];
        for (int i = 0; i < INTN; ++i) lg[i] = gate_b[r * 4 + i];
        for (int d = 0; d < 64; ++d) {
            float xv = x_lds[n * 64 + d] * beh[r * 64 + d];
            for (int i = 0; i < INTN; ++i) lg[i] += xv * gate_W[r * 256 + d * 4 + i];
        }
        float m = lg[0];
        for (int i = 1; i < INTN; ++i) m = fmaxf(m, lg[i]);
        float s = 0.f;
        for (int i = 0; i < INTN; ++i) { lg[i] = __expf(lg[i] - m); s += lg[i]; }
        float inv = 1.f / s;
        for (int i = 0; i < INTN; ++i) g_lds[(r * 16 + n) * 4 + i] = lg[i] * inv;
    }
    __syncthreads();
    int lo = lane & 15, hi = lane >> 4;
    v16h A0, A1;
    #pragma unroll
    for (int h = 0; h < 16; ++h) {
        int k = (h < 8 ? h : h + 8) + hi * 8;
        A0[h] = (_Float16)x_lds[lo * 64 + k];
        A1[h] = (_Float16)x_lds[lo * 64 + 32 + k];
    }
    // per column-tile: intents (WMMA) then gate contraction, store, move on
    #pragma unroll
    for (int et = 0; et < 4; ++et) {
        v8f acc[INTN];
        #pragma unroll
        for (int i = 0; i < INTN; ++i) {
            v8f c = {};
            c = wmma_f16(A0, pk[i * 256 + (et * 2 + 0) * 32 + lane], c);
            c = wmma_f16(A1, pk[i * 256 + (et * 2 + 1) * 32 + lane], c);
            float b = intent_b[i * 64 + et * 16 + lo];
            #pragma unroll
            for (int j = 0; j < 8; ++j) c[j] = lrelu(c[j] + b);
            acc[i] = c;
        }
        #pragma unroll
        for (int r = 0; r < RRR; ++r) {
            #pragma unroll
            for (int j = 0; j < 8; ++j) {
                int m = j + hi * 8;
                float s = 0.f;
                for (int i = 0; i < INTN; ++i) s += acc[i][j] * g_lds[(r * 16 + m) * 4 + i];
                int idx = ((base + m) * 3 + r) * 64 + et * 16 + lo;
                ego[idx] = s;
                all_emb[idx] = s;
            }
        }
    }
}

// ---------------------------------------------------------------------------
// GEMM tile kernel: Y[16x64] (+)= alpha*act((X*scale)@W); optional second
// destination Y2 += same value (fuses the running-sum buffers, saving full
// axpy passes).  4 waves/block, one 16-row tile per wave.  X tile staged to
// LDS via CDNA5 async-to-LDS (global_load_async_to_lds_b128 + s_wait_asynccnt);
// B operands are single 32B per-lane loads from the packed f16 buffer.
// ---------------------------------------------------------------------------
#define ASYNC_LD(OFF)                                                         \
    asm volatile("global_load_async_to_lds_b128 %0, %1, off offset:" #OFF     \
                 :: "v"(ldsOff), "v"(gp) : "memory")

__global__ void gemm64_kernel(const float* __restrict__ X,
                              const v16h* __restrict__ Wpk,
                              const float* __restrict__ scaleVec,
                              float alpha, int doLrelu,
                              float* __restrict__ Y, int yStride, int yOff,
                              int accY, float* __restrict__ Y2) {
    __shared__ float x_lds[4][16 * 64];
    int wave = threadIdx.x >> 5;
    int lane = threadIdx.x & 31;
    int tile = blockIdx.x * 4 + wave;
    if (tile >= NTILE) return;
    int base = tile * 16;

    // async copy of the contiguous 4KB X tile into this wave's LDS slice
    unsigned ldsOff = (unsigned)(uintptr_t)(&x_lds[wave][0]) + (unsigned)(lane * 16);
    const char* gp = (const char*)(X + (size_t)base * 64) + lane * 16;
    ASYNC_LD(0);    ASYNC_LD(512);  ASYNC_LD(1024); ASYNC_LD(1536);
    ASYNC_LD(2048); ASYNC_LD(2560); ASYNC_LD(3072); ASYNC_LD(3584);
    asm volatile("s_wait_asynccnt 0x0" ::: "memory");

    int lo = lane & 15, hi = lane >> 4;
    const float* xr = &x_lds[wave][lo * 64];
    v16h A0, A1;
    if (scaleVec) {
        #pragma unroll
        for (int h = 0; h < 16; ++h) {
            int k = (h < 8 ? h : h + 8) + hi * 8;
            A0[h] = (_Float16)(xr[k] * scaleVec[k]);
            A1[h] = (_Float16)(xr[32 + k] * scaleVec[32 + k]);
        }
    } else {
        #pragma unroll
        for (int h = 0; h < 16; ++h) {
            int k = (h < 8 ? h : h + 8) + hi * 8;
            A0[h] = (_Float16)xr[k];
            A1[h] = (_Float16)xr[32 + k];
        }
    }
    #pragma unroll
    for (int et = 0; et < 4; ++et) {
        int e = et * 16 + lo;
        v8f c = {};
        c = wmma_f16(A0, Wpk[(et * 2 + 0) * 32 + lane], c);
        c = wmma_f16(A1, Wpk[(et * 2 + 1) * 32 + lane], c);
        #pragma unroll
        for (int j = 0; j < 8; ++j) {
            int m = j + hi * 8;
            float v = c[j];
            if (doLrelu) v = lrelu(v);
            v *= alpha;
            size_t idx = (size_t)(base + m) * yStride + yOff + e;
            if (accY) Y[idx] += v; else Y[idx] = v;
            if (Y2)   Y2[idx] += v;
        }
    }
}

// ---------------------------------------------------------------------------
// SpMM: out[row] += val * x[col]; one wave per edge, 2 dims per lane,
// L2-resident global_atomic_add_f32 scatter.
// ---------------------------------------------------------------------------
__global__ void spmm_kernel(const int* __restrict__ rows,
                            const int* __restrict__ cols,
                            const float* __restrict__ vals,
                            const float* __restrict__ x,
                            int xRowStride, int xOff,
                            float* __restrict__ out) {
    int e = blockIdx.x * 8 + (threadIdx.x >> 5);
    if (e >= EDG) return;
    int lane = threadIdx.x & 31;
    int r = rows[e], c = cols[e];
    float v = vals[e];
    const float* xp = x + (size_t)c * xRowStride + xOff + lane * 2;
    float2 xv = *(const float2*)xp;
    float* op = out + (size_t)r * 64 + lane * 2;
    unsafeAtomicAdd(op,     xv.x * v);
    unsafeAtomicAdd(op + 1, xv.y * v);
}

__global__ void zero_kernel(float4* __restrict__ p, int n4) {
    int i = blockIdx.x * blockDim.x + threadIdx.x;
    if (i < n4) p[i] = make_float4(0.f, 0.f, 0.f, 0.f);
}

__global__ void rela_update_kernel(const float* __restrict__ rin,
                                   const float* __restrict__ W,
                                   float* __restrict__ rout) {
    int t = threadIdx.x;            // 192
    int r = t >> 6, e = t & 63;
    float s = 0.f;
    for (int d = 0; d < 64; ++d) s += rin[r * 64 + d] * W[d * 64 + e];
    rout[t] = s;
}

// Per-node attention softmax over r, combine, write all_out rows.
__global__ void attention_kernel(const float* __restrict__ allg,
                                 const float* __restrict__ alle,
                                 const float* __restrict__ u,
                                 float* __restrict__ out) {
    int n = blockIdx.x * blockDim.x + threadIdx.x;
    if (n >= NTOT) return;
    const float* up = u + (size_t)n * 64;
    const float* ep = alle + (size_t)n * 192;
    const float* gp = allg + (size_t)n * 64;
    float lg[3];
    for (int r = 0; r < 3; ++r) {
        float s = 0.f;
        for (int e = 0; e < 64; ++e) s += ep[r * 64 + e] * up[e];
        lg[r] = s;                  // u already carries the 1/9 scaling
    }
    float m = fmaxf(lg[0], fmaxf(lg[1], lg[2]));
    float a0 = __expf(lg[0] - m), a1 = __expf(lg[1] - m), a2 = __expf(lg[2] - m);
    float inv = 1.f / (a0 + a1 + a2);
    float att[3] = {a0 * inv, a1 * inv, a2 * inv};
    const float inv3 = 1.f / 3.f;   // 1/(K+1)
    float* op = out + (size_t)n * 256;
    for (int e = 0; e < 64; ++e) {
        float g = gp[e] * inv3;
        for (int r = 0; r < 3; ++r)
            op[r * 64 + e] = g * att[r] + (1.f - att[r]) * (ep[r * 64 + e] * inv3);
        op[192 + e] = g;
    }
}

// Token row (zeros) + rela_final = mean(rela0..2)
__global__ void finalize_kernel(const float* __restrict__ rela, float* __restrict__ out) {
    int t = threadIdx.x;            // 256
    out[(size_t)NTOT * 256 + t] = 0.f;
    if (t < 192)
        out[(size_t)NTOT * 256 + 256 + t] =
            (rela[t] + rela[192 + t] + rela[384 + t]) * (1.f / 3.f);
}

// ---------------------------------------------------------------------------
extern "C" void kernel_launch(void* const* d_in, const int* in_sizes, int n_in,
                              void* d_out, int out_size, void* d_ws, size_t ws_size,
                              hipStream_t stream) {
    (void)in_sizes; (void)n_in; (void)out_size; (void)ws_size;
    const float* uemb     = (const float*)d_in[0];
    const float* iemb     = (const float*)d_in[1];
    const float* beh      = (const float*)d_in[2];
    const float* W_gcn    = (const float*)d_in[3];
    const float* W_beh    = (const float*)d_in[4];
    const float* W_att1   = (const float*)d_in[5];
    const float* W_att2   = (const float*)d_in[6];
    const float* intent_W = (const float*)d_in[7];
    const float* intent_b = (const float*)d_in[8];
    const float* gate_W   = (const float*)d_in[9];
    const float* gate_b   = (const float*)d_in[10];
    const float* adj_vals = (const float*)d_in[11];
    const int*   adj_rows = (const int*)d_in[12];
    const int*   adj_cols = (const int*)d_in[13];
    float* out = (float*)d_out;
    float* ws  = (float*)d_ws;

    const size_t nf64  = (size_t)NTOT * 64;
    const size_t nf192 = (size_t)NTOT * 192;
    float* gA   = ws;
    float* gB   = gA   + nf64;
    float* allg = gB   + nf64;
    float* egoA = allg + nf64;
    float* egoB = egoA + nf192;
    float* alle = egoB + nf192;
    float* tmp  = alle + nf192;
    float* rela = tmp  + nf64;              // 3 * 192 floats
    _Float16* pk = (_Float16*)(rela + 576); // 8 matrices * 4096 halfs (32B aligned)

    hipMemcpyAsync(rela, beh, 192 * sizeof(float), hipMemcpyDeviceToDevice, stream);

    pack_weights_kernel<<<8, 256, 0, stream>>>(intent_W, W_gcn, W_att1, W_att2, pk);

    fused_init_kernel<<<NTILE, 32, 0, stream>>>(uemb, iemb, beh, (const v16h*)pk,
                                                intent_b, gate_W, gate_b,
                                                gA, allg, egoA, alle);

    const int z64  = (int)(nf64 / 4),  zb64  = (z64 + 255) / 256;
    const int spmmBlocks = EDG / 8;
    const int gemmBlocks = (NTILE + 3) / 4;

    float* gSrc = gA;  float* gDst = gB;
    float* eSrc = egoA; float* eDst = egoB;
    for (int k = 0; k < KLAY; ++k) {
        const v16h* Wgpk  = (const v16h*)(pk + (size_t)(4 + k) * 4096);
        const float* relk = rela + k * 192;
        // global path: gDst = sum_r coef*lrelu(...); allg += same (fused Y2)
        for (int r = 0; r < RRR; ++r) {
            zero_kernel<<<zb64, 256, 0, stream>>>((float4*)tmp, z64);
            spmm_kernel<<<spmmBlocks, 256, 0, stream>>>(adj_rows + (size_t)r * EDG,
                                                        adj_cols + (size_t)r * EDG,
                                                        adj_vals + (size_t)r * EDG,
                                                        gSrc, 64, 0, tmp);
            gemm64_kernel<<<gemmBlocks, 128, 0, stream>>>(tmp, Wgpk, relk + r * 64,
                                                          1.f / 3.f, 1, gDst, 64, 0,
                                                          (r > 0) ? 1 : 0, allg);
        }
        // ego path: eDst[:,r,:] = lrelu(...); alle += same (fused Y2)
        for (int r = 0; r < RRR; ++r) {
            zero_kernel<<<zb64, 256, 0, stream>>>((float4*)tmp, z64);
            spmm_kernel<<<spmmBlocks, 256, 0, stream>>>(adj_rows + (size_t)r * EDG,
                                                        adj_cols + (size_t)r * EDG,
                                                        adj_vals + (size_t)r * EDG,
                                                        eSrc, 192, r * 64, tmp);
            gemm64_kernel<<<gemmBlocks, 128, 0, stream>>>(tmp, Wgpk, relk + r * 64,
                                                          1.f, 1, eDst, 192, r * 64,
                                                          0, alle);
        }
        rela_update_kernel<<<1, 192, 0, stream>>>(relk, W_beh + k * 4096, rela + (k + 1) * 192);
        float* t;
        t = gSrc; gSrc = gDst; gDst = t;
        t = eSrc; eSrc = eDst; eDst = t;
    }

    // attention: t1 = all_global @ W_att1 ; u = t1 @ W_att2^T * (1/9)
    float* t1 = gDst;   // free ping-pong buffer
    gemm64_kernel<<<gemmBlocks, 128, 0, stream>>>(allg, (const v16h*)(pk + (size_t)6 * 4096),
                                                  nullptr, 1.f, 0, t1, 64, 0, 0, nullptr);
    gemm64_kernel<<<gemmBlocks, 128, 0, stream>>>(t1, (const v16h*)(pk + (size_t)7 * 4096),
                                                  nullptr, 1.f / 9.f, 0, tmp, 64, 0, 0, nullptr);

    attention_kernel<<<(NTOT + 255) / 256, 256, 0, stream>>>(allg, alle, tmp, out);
    finalize_kernel<<<1, 256, 0, stream>>>(rela, out);
}
